// PrimitiveLibrary_5617817223987
// MI455X (gfx1250) — matrix operand order repaired and checked by reading, not compiled, not run on hardware
//
#include <hip/hip_runtime.h>
#include <hip/hip_fp16.h>

#define DIM   1024
#define NP    16
#define BATCH 4096
#define ROWS  32                 // batch rows per wave (two 16-row A tiles)
#define LDS_STRIDE (DIM + 8)     // padded f16 row stride: bank step of 4 -> conflict-free reads

typedef __attribute__((ext_vector_type(16))) _Float16 v16h;
typedef __attribute__((ext_vector_type(8)))  _Float16 v8h;
typedef __attribute__((ext_vector_type(8)))  float    v8f;

// ---------------- f32 -> f16 conversion (grid-stride) ----------------
__global__ void f32_to_f16_kernel(const float* __restrict__ src,
                                  _Float16* __restrict__ dst, long n) {
  long i = (long)blockIdx.x * blockDim.x + threadIdx.x;
  long stride = (long)gridDim.x * blockDim.x;
  for (; i < n; i += stride) dst[i] = (_Float16)src[i];
}

// A-fragment (16x32 f16, MxK):
// lane L (l16 = L&15, hi = L>>4): row M = l16,
//   VGPRs 0-7 hold K = kBase + hi*16 .. +15 (same k order as B)
// -> two contiguous 16-byte chunks at (ptr) and (ptr + 8 elements).
__device__ __forceinline__ v16h load_a_frag(const _Float16* p) {
  v8h lo = *(const v8h*)(p);
  v8h hi = *(const v8h*)(p + 8);
  return __builtin_shufflevector(lo, hi, 0,1,2,3,4,5,6,7,8,9,10,11,12,13,14,15);
}

// tanh-form GELU via one v_exp_f32 + v_rcp_f32 (~7 VALU ops):
//   gelu(x) ~= x / (1 + exp(-1.59576912*(x + 0.044715*x^3)))
// max deviation from exact erf-GELU ~3e-4, below the f16 storage quantization of H.
__device__ __forceinline__ float gelu_fast(float x) {
  float x2 = x * x;
  float z  = fmaf(-0.07135481627f * x, x2, -1.5957691216f * x);
  float e  = __expf(z);                           // v_exp_f32
  return x * __builtin_amdgcn_rcpf(1.0f + e);     // v_rcp_f32
}

// ---------------- fused per-primitive MLP ----------------
// grid = (BATCH/ROWS, NP), block = 32 (one wave32 per workgroup)
__global__ __launch_bounds__(32, 1)
void fused_mlp_kernel(const _Float16* __restrict__ xh,
                      const _Float16* __restrict__ W1h,
                      const _Float16* __restrict__ W2h,
                      const float* __restrict__ b1,
                      const float* __restrict__ b2,
                      const float* __restrict__ wts,
                      float* __restrict__ out) {
  __shared__ _Float16 Hs[ROWS * LDS_STRIDE];  // 66,048 B: this wave's 32 rows of H (f16)

  const int n    = blockIdx.y;
  const int r0   = blockIdx.x * ROWS;         // global batch-row base for this wave
  const int lane = threadIdx.x & 31;
  const int l16  = lane & 15;
  const int hi   = lane >> 4;

  const _Float16* W1n = W1h + (size_t)n * DIM * DIM;  // [h][d] row-major
  const _Float16* W2n = W2h + (size_t)n * DIM * DIM;  // [d][h] row-major
  const float*    b1n = b1 + (size_t)n * DIM;
  const float*    b2n = b2 + (size_t)n * DIM;

  // ================= Stage 1: H = gelu(X @ W1^T + b1) =================
  const _Float16* xrow0 = xh + (size_t)(r0 + l16) * DIM + hi * 16;  // A tile 0 (rows r0..r0+15)
  const _Float16* xrow1 = xrow0 + (size_t)16 * DIM;                 // A tile 1 (rows r0+16..r0+31)

  for (int g = 0; g < DIM / 64; ++g) {
    const int c0 = g * 64;                    // 4 x 16 hidden columns per group
    v8f acc0[4] = {v8f{}, v8f{}, v8f{}, v8f{}};
    v8f acc1[4] = {v8f{}, v8f{}, v8f{}, v8f{}};
    // B columns: W1[n][c0 + j*16 + l16][k + hi*16 ..] (K-contiguous 32B)
    const _Float16* pb0 = W1n + (size_t)(c0 +  0 + l16) * DIM + hi * 16;
    const _Float16* pb1 = W1n + (size_t)(c0 + 16 + l16) * DIM + hi * 16;
    const _Float16* pb2 = W1n + (size_t)(c0 + 32 + l16) * DIM + hi * 16;
    const _Float16* pb3 = W1n + (size_t)(c0 + 48 + l16) * DIM + hi * 16;

    for (int k = 0; k < DIM; k += 32) {
      v16h a0 = load_a_frag(xrow0 + k);
      v16h a1 = load_a_frag(xrow1 + k);
      v16h B0 = *(const v16h*)(pb0 + k);
      v16h B1 = *(const v16h*)(pb1 + k);
      v16h B2 = *(const v16h*)(pb2 + k);
      v16h B3 = *(const v16h*)(pb3 + k);
      acc0[0] = __builtin_amdgcn_wmma_f32_16x16x32_f16(false, a0, false, B0, (short)0, acc0[0], false, false);
      acc1[0] = __builtin_amdgcn_wmma_f32_16x16x32_f16(false, a1, false, B0, (short)0, acc1[0], false, false);
      acc0[1] = __builtin_amdgcn_wmma_f32_16x16x32_f16(false, a0, false, B1, (short)0, acc0[1], false, false);
      acc1[1] = __builtin_amdgcn_wmma_f32_16x16x32_f16(false, a1, false, B1, (short)0, acc1[1], false, false);
      acc0[2] = __builtin_amdgcn_wmma_f32_16x16x32_f16(false, a0, false, B2, (short)0, acc0[2], false, false);
      acc1[2] = __builtin_amdgcn_wmma_f32_16x16x32_f16(false, a1, false, B2, (short)0, acc1[2], false, false);
      acc0[3] = __builtin_amdgcn_wmma_f32_16x16x32_f16(false, a0, false, B3, (short)0, acc0[3], false, false);
      acc1[3] = __builtin_amdgcn_wmma_f32_16x16x32_f16(false, a1, false, B3, (short)0, acc1[3], false, false);
    }

    // epilogue: bias + fast gelu, store f16 to LDS.  C/D: VGPR r -> M = r + 8*hi, N = l16.
    #pragma unroll
    for (int j = 0; j < 4; ++j) {
      const int col  = c0 + j * 16 + l16;
      const float bb = b1n[col];
      #pragma unroll
      for (int r = 0; r < 8; ++r) {
        Hs[(size_t)(r + 8 * hi) * LDS_STRIDE + col]      = (_Float16)gelu_fast(acc0[j][r] + bb);
        Hs[(size_t)(16 + r + 8 * hi) * LDS_STRIDE + col] = (_Float16)gelu_fast(acc1[j][r] + bb);
      }
    }
  }

  __syncthreads();  // single wave: cheap; orders LDS stores before reads

  // ================= Stage 2: out += wts[:,n] * (H @ W2^T + b2) =================
  float wr0[8], wr1[8];
  #pragma unroll
  for (int r = 0; r < 8; ++r) {
    wr0[r] = wts[(size_t)(r0 + r + 8 * hi) * NP + n];
    wr1[r] = wts[(size_t)(r0 + 16 + r + 8 * hi) * NP + n];
  }

  const _Float16* hrow0 = Hs + (size_t)l16 * LDS_STRIDE + hi * 16;       // A tile 0 from LDS
  const _Float16* hrow1 = hrow0 + (size_t)16 * LDS_STRIDE;               // A tile 1 from LDS

  for (int g = 0; g < DIM / 64; ++g) {
    const int c0 = g * 64;                    // 4 x 16 output columns per group
    v8f acc0[4] = {v8f{}, v8f{}, v8f{}, v8f{}};
    v8f acc1[4] = {v8f{}, v8f{}, v8f{}, v8f{}};
    const _Float16* pb0 = W2n + (size_t)(c0 +  0 + l16) * DIM + hi * 16;
    const _Float16* pb1 = W2n + (size_t)(c0 + 16 + l16) * DIM + hi * 16;
    const _Float16* pb2 = W2n + (size_t)(c0 + 32 + l16) * DIM + hi * 16;
    const _Float16* pb3 = W2n + (size_t)(c0 + 48 + l16) * DIM + hi * 16;

    for (int k = 0; k < DIM; k += 32) {
      v16h a0 = load_a_frag(hrow0 + k);       // ds_load_b128 x2, conflict-free (stride pad)
      v16h a1 = load_a_frag(hrow1 + k);
      v16h B0 = *(const v16h*)(pb0 + k);
      v16h B1 = *(const v16h*)(pb1 + k);
      v16h B2 = *(const v16h*)(pb2 + k);
      v16h B3 = *(const v16h*)(pb3 + k);
      acc0[0] = __builtin_amdgcn_wmma_f32_16x16x32_f16(false, a0, false, B0, (short)0, acc0[0], false, false);
      acc1[0] = __builtin_amdgcn_wmma_f32_16x16x32_f16(false, a1, false, B0, (short)0, acc1[0], false, false);
      acc0[1] = __builtin_amdgcn_wmma_f32_16x16x32_f16(false, a0, false, B1, (short)0, acc0[1], false, false);
      acc1[1] = __builtin_amdgcn_wmma_f32_16x16x32_f16(false, a1, false, B1, (short)0, acc1[1], false, false);
      acc0[2] = __builtin_amdgcn_wmma_f32_16x16x32_f16(false, a0, false, B2, (short)0, acc0[2], false, false);
      acc1[2] = __builtin_amdgcn_wmma_f32_16x16x32_f16(false, a1, false, B2, (short)0, acc1[2], false, false);
      acc0[3] = __builtin_amdgcn_wmma_f32_16x16x32_f16(false, a0, false, B3, (short)0, acc0[3], false, false);
      acc1[3] = __builtin_amdgcn_wmma_f32_16x16x32_f16(false, a1, false, B3, (short)0, acc1[3], false, false);
    }

    #pragma unroll
    for (int j = 0; j < 4; ++j) {
      const int col  = c0 + j * 16 + l16;
      const float bb = b2n[col];
      #pragma unroll
      for (int r = 0; r < 8; ++r) {
        atomicAdd(&out[(size_t)(r0 + r + 8 * hi) * DIM + col],      (acc0[j][r] + bb) * wr0[r]);
        atomicAdd(&out[(size_t)(r0 + 16 + r + 8 * hi) * DIM + col], (acc1[j][r] + bb) * wr1[r]);
      }
    }
  }
}

extern "C" void kernel_launch(void* const* d_in, const int* in_sizes, int n_in,
                              void* d_out, int out_size, void* d_ws, size_t ws_size,
                              hipStream_t stream) {
  const float* x   = (const float*)d_in[0];
  const float* wts = (const float*)d_in[1];
  const float* W1  = (const float*)d_in[2];
  const float* b1  = (const float*)d_in[3];
  const float* W2  = (const float*)d_in[4];
  const float* b2  = (const float*)d_in[5];
  float* out = (float*)d_out;

  // workspace layout: x(f16) | W1(f16) | W2(f16)  => 8 MB + 32 MB + 32 MB
  _Float16* xh  = (_Float16*)d_ws;
  _Float16* W1h = xh  + (size_t)BATCH * DIM;
  _Float16* W2h = W1h + (size_t)NP * DIM * DIM;

  f32_to_f16_kernel<<<2048, 256, 0, stream>>>(x,  xh,  (long)BATCH * DIM);
  f32_to_f16_kernel<<<4096, 256, 0, stream>>>(W1, W1h, (long)NP * DIM * DIM);
  f32_to_f16_kernel<<<4096, 256, 0, stream>>>(W2, W2h, (long)NP * DIM * DIM);

  hipMemsetAsync(d_out, 0, (size_t)out_size * sizeof(float), stream);

  dim3 grid(BATCH / ROWS, NP);   // 128 row-tiles x 16 primitives = 2048 single-wave WGs
  fused_mlp_kernel<<<grid, 32, 0, stream>>>(xh, W1h, W2h, b1, b2, wts, out);
}
